// AdvancedMultiHeadAttention_72859825209716
// MI455X (gfx1250) — matrix operand order repaired and checked by reading, not compiled, not run on hardware
//
#include <hip/hip_runtime.h>
#include <hip/hip_bf16.h>

typedef __bf16 bf16_t;
typedef __attribute__((ext_vector_type(16))) __bf16 v16bf;
typedef __attribute__((ext_vector_type(8)))  float  v8f;

#define DIM 1024
#define NHEADS 16
#define HDIM 64
#define SSEQ 2048
#define BATCH 2
#define MAXDIST 128

__device__ __forceinline__ int imin(int a, int b) { return a < b ? a : b; }
__device__ __forceinline__ int imax(int a, int b) { return a > b ? a : b; }

__device__ __forceinline__ v8f wmma_bf16(v16bf a, v16bf b, v8f c) {
  return __builtin_amdgcn_wmma_f32_16x16x32_bf16(false, a, false, b, (short)0, c,
                                                 false, false);
}

// Load 16 contiguous bf16 (B-matrix fragment chunk): 32 bytes, 16B-aligned.
__device__ __forceinline__ v16bf load16(const bf16_t* p) {
  union { uint4 u[2]; v16bf v; } t;
  t.u[0] = ((const uint4*)p)[0];
  t.u[1] = ((const uint4*)p)[1];
  return t.v;
}

// A-matrix fragment chunk: 8 bf16 at p and 8 bf16 at p+16 (16-bit A layout:
// lane half selects K sub-blocks {half*8..half*8+7, 16+half*8..16+half*8+7}).
__device__ __forceinline__ v16bf load8x2(const bf16_t* p) {
  union { uint4 u[2]; v16bf v; } t;
  t.u[0] = *(const uint4*)p;
  t.u[1] = *(const uint4*)(p + 16);
  return t.v;
}

// Same A-fragment pattern but converting from fp32 source.
__device__ __forceinline__ v16bf cvt8x2(const float* p) {
  union { float4 f4[2]; float f[8]; } a, b;
  a.f4[0] = ((const float4*)p)[0];
  a.f4[1] = ((const float4*)p)[1];
  b.f4[0] = ((const float4*)(p + 16))[0];
  b.f4[1] = ((const float4*)(p + 16))[1];
  v16bf v;
#pragma unroll
  for (int j = 0; j < 8; ++j) {
    v[j]     = (bf16_t)a.f[j];
    v[j + 8] = (bf16_t)b.f[j];
  }
  return v;
}

// ---------------------------------------------------------------------------
// W [K=1024][N=1024] fp32 -> WT [N][K] bf16 (so B-fragments are K-contiguous)
// ---------------------------------------------------------------------------
__global__ __launch_bounds__(256) void wt_convert(const float* __restrict__ W,
                                                  bf16_t* __restrict__ WT) {
  __shared__ float tile[32][33];
  const int tx = threadIdx.x & 31;
  const int ty = threadIdx.x >> 5;  // 0..7
  const int n0 = blockIdx.x * 32;
  const int k0 = blockIdx.y * 32;
#pragma unroll
  for (int i = 0; i < 4; ++i)
    tile[ty + i * 8][tx] = W[(size_t)(k0 + ty + i * 8) * DIM + n0 + tx];
  __syncthreads();
#pragma unroll
  for (int i = 0; i < 4; ++i)
    WT[(size_t)(n0 + ty + i * 8) * DIM + k0 + tx] = (bf16_t)tile[tx][ty + i * 8];
}

// ---------------------------------------------------------------------------
// C[4096,1024] = X(f32) @ W + bias, written bf16 in attention layout.
// mode 0: out[((b*H+h)*S+s)*64+d]   (Q, K)
// mode 1: out[((b*H+h)*64+d)*S+s]   (V transposed for P@V B-fragments)
// One wave computes a 16x64 tile; block = 8 waves = 128 rows x 64 cols.
// ---------------------------------------------------------------------------
__global__ __launch_bounds__(256) void proj_gemm_f32(
    const float* __restrict__ X, const bf16_t* __restrict__ WT,
    const float* __restrict__ bias, bf16_t* __restrict__ Out, int mode) {
  const int lane = threadIdx.x & 31;
  const int wave = threadIdx.x >> 5;
  const int half = lane >> 4;
  const int lan  = lane & 15;
  const int m0 = blockIdx.x * 128 + wave * 16;
  const int n0 = blockIdx.y * 64;

  v8f acc[4] = {};
  const float* arow = X + (size_t)(m0 + lan) * DIM;

  for (int k = 0; k < DIM; k += 32) {
    v16bf a = cvt8x2(arow + k + half * 8);
#pragma unroll
    for (int nt = 0; nt < 4; ++nt) {
      const bf16_t* bp =
          WT + (size_t)(n0 + nt * 16 + lan) * DIM + k + half * 8;
      acc[nt] = wmma_bf16(a, load8x2(bp), acc[nt]);
    }
  }
  // f32 C layout: VGPR r -> row r + 8*half, lane&15 -> col
#pragma unroll
  for (int nt = 0; nt < 4; ++nt) {
#pragma unroll
    for (int r = 0; r < 8; ++r) {
      const int mm = m0 + half * 8 + r;
      const int nn = n0 + nt * 16 + lan;
      const float v = acc[nt][r] + bias[nn];
      const int b = mm >> 11;       // S = 2048
      const int s = mm & (SSEQ - 1);
      const int h = nn >> 6;
      const int d = nn & 63;
      size_t idx;
      if (mode == 0)
        idx = ((size_t)(b * NHEADS + h) * SSEQ + s) * HDIM + d;
      else
        idx = ((size_t)(b * NHEADS + h) * HDIM + d) * SSEQ + s;
      Out[idx] = (bf16_t)v;
    }
  }
}

// ---------------------------------------------------------------------------
// Flash attention: one wave per (b, h, 16 query rows), 32 keys per step.
// ---------------------------------------------------------------------------
__global__ __launch_bounds__(128) void attention_kernel(
    const bf16_t* __restrict__ Qh, const bf16_t* __restrict__ Kh,
    const bf16_t* __restrict__ Vt, const unsigned char* __restrict__ kmask,
    const float* __restrict__ logit_scale, const float* __restrict__ bias_table,
    bf16_t* __restrict__ AttnOut) {
  const int lane = threadIdx.x & 31;
  const int wave = threadIdx.x >> 5;  // 0..3
  const int half = lane >> 4;
  const int lan  = lane & 15;
  const int h = blockIdx.y;
  const int b = blockIdx.z;
  const int q0 = (blockIdx.x * 4 + wave) * 16;

  __shared__ __attribute__((aligned(32))) bf16_t ldsP[4][16 * 32];
  bf16_t* P = ldsP[wave];

  const size_t bh = (size_t)(b * NHEADS + h);
  const bf16_t* Qbase = Qh + bh * SSEQ * HDIM;
  const bf16_t* Kbase = Kh + bh * SSEQ * HDIM;
  const bf16_t* Vbase = Vt + bh * HDIM * SSEQ;
  const unsigned char* mrow = kmask + (size_t)b * SSEQ;

  const float scale = 0.125f * fminf(__expf(logit_scale[h]), 4.0f);

  // Q A-fragments (rows q0..q0+15, hd split into two K=32 chunks)
  v16bf aq[2];
  {
    const bf16_t* qrow = Qbase + (size_t)(q0 + lan) * HDIM;
#pragma unroll
    for (int c = 0; c < 2; ++c) aq[c] = load8x2(qrow + c * 32 + half * 8);
  }

  v8f O[4] = {};
  float runm[8], runl[8];
#pragma unroll
  for (int r = 0; r < 8; ++r) { runm[r] = -__builtin_inff(); runl[r] = 0.f; }
  const int mbase = half * 8;
  const float neg = -3.4028234663852886e38f;

  for (int kb = 0; kb < SSEQ; kb += 32) {
    if (kb + 32 < SSEQ) {  // uniform-condition prefetch of next chunk
      __builtin_prefetch(Kbase + (size_t)(kb + 32 + lan) * HDIM, 0, 0);
      __builtin_prefetch(Vbase + (size_t)lan * SSEQ + kb + 32, 0, 0);
    }
    // ---- scores S = Q @ K^T for 2 x 16-key tiles ----
    v8f S0 = {}, S1 = {};
    const bf16_t* krow0 = Kbase + (size_t)(kb + lan) * HDIM;
    const bf16_t* krow1 = Kbase + (size_t)(kb + 16 + lan) * HDIM;
#pragma unroll
    for (int c = 0; c < 2; ++c) {
      S0 = wmma_bf16(aq[c], load8x2(krow0 + c * 32 + half * 8), S0);
      S1 = wmma_bf16(aq[c], load8x2(krow1 + c * 32 + half * 8), S1);
    }
    // ---- scale + relative-position bias + padding mask + online softmax ----
    const int k0pos = kb + lan;
    const int k1pos = kb + 16 + lan;
    const bool m0 = mrow[k0pos] != 0;
    const bool m1 = mrow[k1pos] != 0;
    float p0v[8], p1v[8], alpha[8];
#pragma unroll
    for (int r = 0; r < 8; ++r) {
      const int qpos = q0 + mbase + r;
      const int rel0 = imin(imax(k0pos - qpos, -MAXDIST), MAXDIST) + MAXDIST;
      const int rel1 = imin(imax(k1pos - qpos, -MAXDIST), MAXDIST) + MAXDIST;
      float s0 = S0[r] * scale + bias_table[rel0 * NHEADS + h];
      float s1 = S1[r] * scale + bias_table[rel1 * NHEADS + h];
      if (m0) s0 = neg;
      if (m1) s1 = neg;
      float mx = fmaxf(s0, s1);
#pragma unroll
      for (int d = 1; d < 16; d <<= 1) mx = fmaxf(mx, __shfl_xor(mx, d, 32));
      const float newm = fmaxf(runm[r], mx);
      alpha[r] = __expf(runm[r] - newm);
      p0v[r] = __expf(s0 - newm);
      p1v[r] = __expf(s1 - newm);
      runl[r] = runl[r] * alpha[r] + p0v[r] + p1v[r];
      runm[r] = newm;
    }
#pragma unroll
    for (int nt = 0; nt < 4; ++nt)
#pragma unroll
      for (int r = 0; r < 8; ++r) O[nt][r] *= alpha[r];

    // ---- repack P: f32 C-layout -> bf16 A-layout via LDS ----
    __syncthreads();
#pragma unroll
    for (int r = 0; r < 8; ++r) {
      const int mm = mbase + r;
      P[mm * 32 + lan]      = (bf16_t)p0v[r];
      P[mm * 32 + 16 + lan] = (bf16_t)p1v[r];
    }
    __syncthreads();
    const v16bf ap = load8x2(P + lan * 32 + half * 8);

    // ---- O += P @ V (keys are the K dimension; Vt is key-contiguous) ----
#pragma unroll
    for (int nt = 0; nt < 4; ++nt) {
      const bf16_t* vp =
          Vbase + (size_t)(nt * 16 + lan) * SSEQ + kb + half * 8;
      O[nt] = wmma_bf16(ap, load8x2(vp), O[nt]);
    }
  }

  // normalize rows and write [B,S,DIM] bf16
#pragma unroll
  for (int r = 0; r < 8; ++r) {
    float s = runl[r];
#pragma unroll
    for (int d = 1; d < 16; d <<= 1) s += __shfl_xor(s, d, 32);
    runl[r] = 1.0f / s;
  }
  bf16_t* outbase = AttnOut + (size_t)b * SSEQ * DIM + (size_t)h * HDIM;
#pragma unroll
  for (int nt = 0; nt < 4; ++nt)
#pragma unroll
    for (int r = 0; r < 8; ++r) {
      const int s = q0 + mbase + r;
      outbase[(size_t)s * DIM + nt * 16 + lan] = (bf16_t)(O[nt][r] * runl[r]);
    }
}

// ---------------------------------------------------------------------------
// d_out[4096,1024] f32 = AttnOut(bf16) @ Wo + bo
// ---------------------------------------------------------------------------
__global__ __launch_bounds__(256) void out_gemm_bf16(
    const bf16_t* __restrict__ X, const bf16_t* __restrict__ WT,
    const float* __restrict__ bias, float* __restrict__ Out) {
  const int lane = threadIdx.x & 31;
  const int wave = threadIdx.x >> 5;
  const int half = lane >> 4;
  const int lan  = lane & 15;
  const int m0 = blockIdx.x * 128 + wave * 16;
  const int n0 = blockIdx.y * 64;

  v8f acc[4] = {};
  const bf16_t* arow = X + (size_t)(m0 + lan) * DIM;

  for (int k = 0; k < DIM; k += 32) {
    v16bf a = load8x2(arow + k + half * 8);
#pragma unroll
    for (int nt = 0; nt < 4; ++nt) {
      const bf16_t* bp =
          WT + (size_t)(n0 + nt * 16 + lan) * DIM + k + half * 8;
      acc[nt] = wmma_bf16(a, load8x2(bp), acc[nt]);
    }
  }
#pragma unroll
  for (int nt = 0; nt < 4; ++nt)
#pragma unroll
    for (int r = 0; r < 8; ++r) {
      const int mm = m0 + half * 8 + r;
      const int nn = n0 + nt * 16 + lan;
      Out[(size_t)mm * DIM + nn] = acc[nt][r] + bias[nn];
    }
}

extern "C" void kernel_launch(void* const* d_in, const int* in_sizes, int n_in,
                              void* d_out, int out_size, void* d_ws,
                              size_t ws_size, hipStream_t stream) {
  const float* query = (const float*)d_in[0];
  const float* key   = (const float*)d_in[1];
  const float* value = (const float*)d_in[2];
  const unsigned char* kpm = (const unsigned char*)d_in[3];
  const float* Wq = (const float*)d_in[4];
  const float* bq = (const float*)d_in[5];
  const float* Wk = (const float*)d_in[6];
  const float* bk = (const float*)d_in[7];
  const float* Wv = (const float*)d_in[8];
  const float* bv = (const float*)d_in[9];
  const float* Wo = (const float*)d_in[10];
  const float* bo = (const float*)d_in[11];
  const float* logit_scale = (const float*)d_in[12];
  const float* bias_table  = (const float*)d_in[13];
  float* out = (float*)d_out;

  char* ws = (char*)d_ws;
  const size_t WBYTES = (size_t)DIM * DIM * sizeof(bf16_t);            // 2 MB
  const size_t TBYTES = (size_t)BATCH * NHEADS * SSEQ * HDIM * 2;      // 8 MB
  bf16_t* WTq = (bf16_t*)(ws + 0 * WBYTES);
  bf16_t* WTk = (bf16_t*)(ws + 1 * WBYTES);
  bf16_t* WTv = (bf16_t*)(ws + 2 * WBYTES);
  bf16_t* WTo = (bf16_t*)(ws + 3 * WBYTES);
  bf16_t* Qh  = (bf16_t*)(ws + 4 * WBYTES);
  bf16_t* Kh  = (bf16_t*)(ws + 4 * WBYTES + 1 * TBYTES);
  bf16_t* Vt  = (bf16_t*)(ws + 4 * WBYTES + 2 * TBYTES);
  bf16_t* Ao  = (bf16_t*)(ws + 4 * WBYTES + 3 * TBYTES);

  const dim3 tb(256);
  wt_convert<<<dim3(32, 32), tb, 0, stream>>>(Wq, WTq);
  wt_convert<<<dim3(32, 32), tb, 0, stream>>>(Wk, WTk);
  wt_convert<<<dim3(32, 32), tb, 0, stream>>>(Wv, WTv);
  wt_convert<<<dim3(32, 32), tb, 0, stream>>>(Wo, WTo);

  proj_gemm_f32<<<dim3(32, 16), tb, 0, stream>>>(query, WTq, bq, Qh, 0);
  proj_gemm_f32<<<dim3(32, 16), tb, 0, stream>>>(key,   WTk, bk, Kh, 0);
  proj_gemm_f32<<<dim3(32, 16), tb, 0, stream>>>(value, WTv, bv, Vt, 1);

  attention_kernel<<<dim3(SSEQ / 64, NHEADS, BATCH), dim3(128), 0, stream>>>(
      Qh, Kh, Vt, kpm, logit_scale, bias_table, Ao);

  out_gemm_bf16<<<dim3(32, 16), tb, 0, stream>>>(Ao, WTo, bo, out);
}
